// SpatialEncoder_59090160059031
// MI455X (gfx1250) — matrix operand order repaired and checked by reading, not compiled, not run on hardware
//
#include <hip/hip_runtime.h>
#include <hip/hip_bf16.h>

typedef __attribute__((ext_vector_type(16))) __bf16 v16bf;
typedef __attribute__((ext_vector_type(8)))  float  v8f;

#define Hdim   128
#define Wdim   128
#define Cdim   256
#define Bdim   8
#define TILE_H 8
#define TILE_W 8
#define NPIX   64        // TILE_H * TILE_W pixels per block
#define KTOT   1024      // 4*C reduction dim
#define YPITCH 1032      // halves per pixel row in LDS (1024 + 8 pad -> 4-bank skew)

__device__ __forceinline__ unsigned short f2bf(float f) {
    unsigned int u = __float_as_uint(f);
    u += 0x7FFFu + ((u >> 16) & 1u);   // round-to-nearest-even
    return (unsigned short)(u >> 16);
}

__global__ void wproj_to_bf16(const float* __restrict__ w,
                              unsigned short* __restrict__ o, int n) {
    int i = blockIdx.x * blockDim.x + threadIdx.x;
    if (i < n) o[i] = f2bf(w[i]);
}

union FragU {
    v16bf v;
    uint4 q[2];
};

__global__ __launch_bounds__(256)
void spatial_encoder_fused(const float* __restrict__ x,
                           const float* __restrict__ wh1,
                           const float* __restrict__ wh2,
                           const float* __restrict__ wv1,
                           const float* __restrict__ wv2,
                           const unsigned short* __restrict__ wprojb,
                           float* __restrict__ out) {
    extern __shared__ unsigned short ys[];   // [NPIX][YPITCH] bf16 y-tile

    const int tid = threadIdx.x;
    const int w0  = blockIdx.x * TILE_W;
    const int h0  = blockIdx.y * TILE_H;
    const int b   = blockIdx.z;

    // ------- Phase 1: depthwise convs -> LDS (thread = channel, taps in regs) -------
    {
        const int c = tid;                                  // 0..255
        float h1w[9], h2w[9], v1w[9], v2w[9];
        #pragma unroll
        for (int t = 0; t < 9; ++t) {
            h1w[t] = wh1[c * 9 + t];
            h2w[t] = wh2[c * 9 + t];
            v1w[t] = wv1[c * 9 + t];
            v2w[t] = wv2[c * 9 + t];
        }
        const float* xc = x + ((long)(b * Cdim + c) << 14); // channel plane
        unsigned short* yc = ys + c;                        // k = part*256 + c

        // horizontal 1x9 convs: slide over 16-wide row buffer
        #pragma unroll 1
        for (int ph = 0; ph < TILE_H; ++ph) {
            const float* xr = xc + (h0 + ph) * Wdim + w0;
            float buf[16];
            #pragma unroll
            for (int t = 0; t < 16; ++t) {
                int wc = w0 + t - 4;
                buf[t] = (wc >= 0 && wc < Wdim) ? xr[t - 4] : 0.f;
            }
            #pragma unroll
            for (int pw = 0; pw < TILE_W; ++pw) {
                float a0 = 0.f, a1 = 0.f;
                #pragma unroll
                for (int t = 0; t < 9; ++t) {
                    a0 = fmaf(h1w[t], buf[pw + t], a0);
                    a1 = fmaf(h2w[t], buf[pw + t], a1);
                }
                const int p = ph * TILE_W + pw;
                yc[p * YPITCH + 0]   = f2bf(a0);
                yc[p * YPITCH + 256] = f2bf(a1);
            }
        }
        // vertical 9x1 convs: slide over 16-tall column buffer
        #pragma unroll 1
        for (int pw = 0; pw < TILE_W; ++pw) {
            const float* xcol = xc + (w0 + pw);
            float buf[16];
            #pragma unroll
            for (int t = 0; t < 16; ++t) {
                int hc = h0 + t - 4;
                buf[t] = (hc >= 0 && hc < Hdim) ? xcol[hc * Wdim] : 0.f;
            }
            #pragma unroll
            for (int ph = 0; ph < TILE_H; ++ph) {
                float a2 = 0.f, a3 = 0.f;
                #pragma unroll
                for (int t = 0; t < 9; ++t) {
                    a2 = fmaf(v1w[t], buf[ph + t], a2);
                    a3 = fmaf(v2w[t], buf[ph + t], a3);
                }
                const int p = ph * TILE_W + pw;
                yc[p * YPITCH + 512] = f2bf(a2);
                yc[p * YPITCH + 768] = f2bf(a3);
            }
        }
    }
    __syncthreads();

    // ------- Phase 2: projection GEMM via bf16 WMMA (M=256, N=64, K=1024) -------
    const int wave = tid >> 5;     // 0..7
    const int lane = tid & 31;
    const int lrow = lane & 15;
    const int g    = lane >> 4;    // lane group (K-half select)

    const int mt0 = wave;          // out-ch tiles 16*wave and 16*(wave+8)
    const int mt1 = wave + 8;

    v8f acc0[4] = {};              // [nt] for mt0
    v8f acc1[4] = {};              // [nt] for mt1

    #pragma unroll 2
    for (int kb = 0; kb < 32; ++kb) {
        const int kA = kb * 32 + g * 8;      // A halves 0-7 @ K=g*8, 8-15 @ +16
        const int kB = kb * 32 + g * 8;      // B halves 0-7 @ K=g*8, 8-15 @ +16
        FragU A0, A1, Bf[4];
        const uint4* a0p = (const uint4*)(wprojb + ((mt0 * 16 + lrow) * KTOT + kA));
        const uint4* a1p = (const uint4*)(wprojb + ((mt1 * 16 + lrow) * KTOT + kA));
        A0.q[0] = a0p[0]; A0.q[1] = a0p[2];  // +16 halves
        A1.q[0] = a1p[0]; A1.q[1] = a1p[2];
        #pragma unroll
        for (int nt = 0; nt < 4; ++nt) {
            const uint4* bp = (const uint4*)(ys + (nt * 16 + lrow) * YPITCH + kB);
            Bf[nt].q[0] = bp[0];
            Bf[nt].q[1] = bp[2];
        }
        #pragma unroll
        for (int nt = 0; nt < 4; ++nt) {
            acc0[nt] = __builtin_amdgcn_wmma_f32_16x16x32_bf16(
                false, A0.v, false, Bf[nt].v, (short)0, acc0[nt], false, false);
            acc1[nt] = __builtin_amdgcn_wmma_f32_16x16x32_bf16(
                false, A1.v, false, Bf[nt].v, (short)0, acc1[nt], false, false);
        }
    }

    // ------- Store with ReLU -------
    float* ob = out + (((long)b * Cdim) << 14);
    #pragma unroll
    for (int nt = 0; nt < 4; ++nt) {
        const int p  = nt * 16 + lrow;
        const int ph = p >> 3, pw = p & 7;
        const long off = (long)(h0 + ph) * Wdim + (w0 + pw);
        #pragma unroll
        for (int r = 0; r < 8; ++r) {
            const int m = r + 8 * g;          // C/D layout: M = r + 8*laneGroup
            ob[((long)(mt0 * 16 + m) << 14) + off] = fmaxf(acc0[nt][r], 0.f);
            ob[((long)(mt1 * 16 + m) << 14) + off] = fmaxf(acc1[nt][r], 0.f);
        }
    }
}

extern "C" void kernel_launch(void* const* d_in, const int* in_sizes, int n_in,
                              void* d_out, int out_size, void* d_ws, size_t ws_size,
                              hipStream_t stream) {
    (void)n_in; (void)out_size; (void)ws_size;
    const float* x    = (const float*)d_in[0];
    const float* wh1  = (const float*)d_in[1];
    const float* wh2  = (const float*)d_in[2];
    const float* wv1  = (const float*)d_in[3];
    const float* wv2  = (const float*)d_in[4];
    const float* wprj = (const float*)d_in[5];
    float* out = (float*)d_out;

    // Convert wproj (256 x 1024) to bf16 in workspace (L2-resident, 512 KB)
    unsigned short* wprojb = (unsigned short*)d_ws;
    const int nproj = in_sizes[5];   // 262144
    wproj_to_bf16<<<(nproj + 255) / 256, 256, 0, stream>>>(wprj, wprojb, nproj);

    dim3 grid(Wdim / TILE_W, Hdim / TILE_H, Bdim);   // 16 x 16 x 8 = 2048 blocks
    size_t shmem = (size_t)NPIX * YPITCH * sizeof(unsigned short);  // 132096 B
    spatial_encoder_fused<<<grid, 256, shmem, stream>>>(x, wh1, wh2, wv1, wv2,
                                                        wprojb, out);
}
